// DKEPooling_6605659701699
// MI455X (gfx1250) — matrix operand order repaired and checked by reading, not compiled, not run on hardware
//
#include <hip/hip_runtime.h>
#include <math.h>

#define DD   256
#define NPG  512
#define NB   128
#define ITERN 5

typedef __attribute__((ext_vector_type(16))) _Float16 v16h;
typedef __attribute__((ext_vector_type(8)))  float    v8f;

struct __align__(16) F4 { float a, b, c, d; };          // POD 16B block
union  U16B { F4 f; _Float16 h[8]; };                    // 8 halves view
union  U4B  { unsigned int w; _Float16 h[2]; };          // 2 halves view

// ---- WMMA fragment loaders (wave32 layouts, ISA 7.12.2), ldk = 32 halves --
// A 16x32 f16: lane&15=M; VGPR v<4 -> K=2v,2v+1 ; v>=4 -> K=16+2(v-4); +8 if lane>=16
__device__ __forceinline__ v16h load_a_frag(const _Float16* smem, int row_base,
                                            int lane) {
  const unsigned int* p =
      (const unsigned int*)(smem + (row_base + (lane & 15)) * 32);
  const int khi = (lane & 16) ? 8 : 0;
  union { v16h v; unsigned int u[8]; } r;
#pragma unroll
  for (int i = 0; i < 8; ++i) {
    int k = ((i < 4) ? (2 * i) : (16 + 2 * (i - 4))) + khi;
    r.u[i] = p[k >> 1];
  }
  return r.v;
}
// B 32x16 f16: lane&15=N; VGPR v<4 -> K=2v,2v+1 ; v>=4 -> K=16+2(v-4); +8 if lane>=16
__device__ __forceinline__ v16h load_b_frag(const _Float16* smem, int n_base,
                                            int lane) {
  const unsigned int* p =
      (const unsigned int*)(smem + (n_base + (lane & 15)) * 32);
  const int kh = (lane & 16) ? 8 : 0;
  union { v16h v; unsigned int u[8]; } r;
#pragma unroll
  for (int i = 0; i < 8; ++i) r.u[i] = p[((i < 4) ? i : i + 4) + (kh >> 1)];
  return r.v;
}

// 8 WMMAs: 2 row-tiles x 4 col-tiles on one 32-deep K chunk
__device__ __forceinline__ void mma_step(const _Float16* As, const _Float16* Bs,
                                         int wave, int lane, v8f acc[2][4]) {
  v16h a0 = load_a_frag(As, wave * 32, lane);
  v16h a1 = load_a_frag(As, wave * 32 + 16, lane);
#pragma unroll
  for (int tc = 0; tc < 4; ++tc) {
    v16h bf = load_b_frag(Bs, tc * 16, lane);
    acc[0][tc] = __builtin_amdgcn_wmma_f32_16x16x32_f16(
        false, a0, false, bf, (short)0, acc[0][tc], false, false);
    acc[1][tc] = __builtin_amdgcn_wmma_f32_16x16x32_f16(
        false, a1, false, bf, (short)0, acc[1][tc], false, false);
  }
}

// ---- K1: mean, diff(f16), Frobenius^2 per batch (pure HBM streaming) ------
__global__ __launch_bounds__(256) void prep_kernel(
    const float* __restrict__ feat, const float* __restrict__ noise,
    float* __restrict__ mean, _Float16* __restrict__ Df,
    float* __restrict__ invTrace, float* __restrict__ sqrtNormA) {
  const int b = blockIdx.x;
  const int c = threadIdx.x;
  const size_t base = (size_t)b * NPG * DD;
  float s = 0.f;
  for (int n = 0; n < NPG; ++n) {
    size_t idx = base + (size_t)n * DD + c;
    __builtin_prefetch(feat + idx + 16 * DD, 0, 0);   // global_prefetch_b8
    __builtin_prefetch(noise + idx + 16 * DD, 0, 0);
    s += feat[idx] + 0.01f * noise[idx];
  }
  const float mu = s * (1.0f / NPG);
  mean[b * DD + c] = mu;
  float sq = 0.f;
  for (int n = 0; n < NPG; ++n) {
    size_t idx = base + (size_t)n * DD + c;
    float df = feat[idx] + 0.01f * noise[idx] - mu;
    Df[idx] = (_Float16)df;
    sq += df * df;
  }
  __shared__ float red[256];
  red[c] = sq;
  __syncthreads();
  for (int s2 = 128; s2 > 0; s2 >>= 1) {
    if (c < s2) red[c] += red[c + s2];
    __syncthreads();
  }
  if (c == 0) {
    float frob2 = red[0];                 // trace(D^T D) = 511 * normA
    invTrace[b] = 1.0f / frob2;           // A = G / frob2 (511 cancels)
    sqrtNormA[b] = sqrtf(frob2 * (1.0f / (NPG - 1)));
  }
}

// ---- K2: G = D^T D (K=512), double-buffered; epilogue A, T0=0.5(3I-A) -----
__global__ __launch_bounds__(256) void cov_kernel(
    const _Float16* __restrict__ Df, const float* __restrict__ invTrace,
    _Float16* __restrict__ A16, _Float16* __restrict__ T16) {
  const int b = blockIdx.y;
  const int c0 = blockIdx.x * 64;
  const int tid = threadIdx.x;
  const int lane = tid & 31;
  const int wave = tid >> 5;
  __shared__ __align__(16) _Float16 As[2][DD * 32];
  __shared__ __align__(16) _Float16 Bs[2][64 * 32];
  const _Float16* Db = Df + (size_t)b * NPG * DD;

  // A-stage (transpose): thread owns rows {r2,r2+1} x cols [cb,cb+16)
  const int r2 = (tid & 15) * 2;
  const int cb = (tid >> 4) * 16;
  // B-stage: thread owns cols {n2,n2+1} at rows kkb+8j
  const int n2 = (tid & 31) * 2;
  const int kkb = tid >> 5;

  U16B aReg[2][2];
  unsigned int bReg[4];

  auto gload = [&](int k0) {
#pragma unroll
    for (int rr = 0; rr < 2; ++rr)
#pragma unroll
      for (int i = 0; i < 2; ++i)
        aReg[rr][i].f =
            *(const F4*)(Db + (size_t)(k0 + r2 + rr) * DD + cb + i * 8);
#pragma unroll
    for (int j = 0; j < 4; ++j)
      bReg[j] = *(const unsigned int*)(Db + (size_t)(k0 + kkb + j * 8) * DD +
                                       c0 + n2);
  };
  auto sstore = [&](int buf) {
#pragma unroll
    for (int j = 0; j < 16; ++j) {  // conflict-free transposed b32 stores
      U4B u;
      u.h[0] = aReg[0][j >> 3].h[j & 7];
      u.h[1] = aReg[1][j >> 3].h[j & 7];
      *(unsigned int*)(&As[buf][(cb + j) * 32 + r2]) = u.w;
    }
#pragma unroll
    for (int j = 0; j < 4; ++j) {
      U4B u; u.w = bReg[j];
      int kk = kkb + j * 8;
      Bs[buf][n2 * 32 + kk] = u.h[0];
      Bs[buf][(n2 + 1) * 32 + kk] = u.h[1];
    }
  };

  v8f acc[2][4] = {};
  gload(0);
  sstore(0);
  __syncthreads();
  const int NC = NPG / 32;  // 16 chunks
  for (int kc = 0; kc < NC; ++kc) {
    const int cur = kc & 1;
    if (kc + 1 < NC) gload((kc + 1) * 32);
    mma_step(As[cur], Bs[cur], wave, lane, acc);
    if (kc + 1 < NC) {
      sstore(cur ^ 1);
      __syncthreads();
    }
  }
  const float inv = invTrace[b];
  _Float16* Ab = A16 + (size_t)b * DD * DD;
  _Float16* Tb = T16 + (size_t)b * DD * DD;
  const int nloc = lane & 15;
  const int mhi = (lane & 16) ? 8 : 0;
#pragma unroll
  for (int tr = 0; tr < 2; ++tr)
#pragma unroll
    for (int tc = 0; tc < 4; ++tc) {
      int ng = c0 + tc * 16 + nloc;
#pragma unroll
      for (int r = 0; r < 8; ++r) {
        int mg = wave * 32 + tr * 16 + mhi + r;
        float a = acc[tr][tc][r] * inv;
        Ab[mg * DD + ng] = (_Float16)a;
        Tb[mg * DD + ng] = (_Float16)(0.5f * (((mg == ng) ? 3.0f : 0.0f) - a));
      }
    }
}

// ---- K3: 256x256x256 f16 GEMM, double-buffered; mode1 => C=0.5(3I-P*Q) ----
__global__ __launch_bounds__(256) void ns_gemm_kernel(
    const _Float16* __restrict__ P, const _Float16* __restrict__ Q,
    _Float16* __restrict__ C, int mode) {
  const int b = blockIdx.y;
  const int c0 = blockIdx.x * 64;
  const int tid = threadIdx.x;
  const int lane = tid & 31;
  const int wave = tid >> 5;
  __shared__ __align__(16) _Float16 As[2][DD * 32];
  __shared__ __align__(16) _Float16 Bs[2][64 * 32];
  const _Float16* Pb = P + (size_t)b * DD * DD;
  const _Float16* Qb = Q + (size_t)b * DD * DD;

  const int n2 = (tid & 31) * 2;
  const int kkb = tid >> 5;

  F4 aReg[4];
  unsigned int bReg[4];

  auto gload = [&](int k0) {
    const F4* s = (const F4*)(Pb + (size_t)tid * DD + k0);
#pragma unroll
    for (int i = 0; i < 4; ++i) aReg[i] = s[i];
#pragma unroll
    for (int j = 0; j < 4; ++j)
      bReg[j] = *(const unsigned int*)(Qb + (size_t)(k0 + kkb + j * 8) * DD +
                                       c0 + n2);
  };
  auto sstore = [&](int buf) {
    F4* d = (F4*)(As[buf] + tid * 32);
#pragma unroll
    for (int i = 0; i < 4; ++i) d[i] = aReg[i];
#pragma unroll
    for (int j = 0; j < 4; ++j) {
      U4B u; u.w = bReg[j];
      int kk = kkb + j * 8;
      Bs[buf][n2 * 32 + kk] = u.h[0];
      Bs[buf][(n2 + 1) * 32 + kk] = u.h[1];
    }
  };

  v8f acc[2][4] = {};
  gload(0);
  sstore(0);
  __syncthreads();
  const int NC = DD / 32;  // 8 chunks
  for (int kc = 0; kc < NC; ++kc) {
    const int cur = kc & 1;
    if (kc + 1 < NC) gload((kc + 1) * 32);
    mma_step(As[cur], Bs[cur], wave, lane, acc);
    if (kc + 1 < NC) {
      sstore(cur ^ 1);
      __syncthreads();
    }
  }
  _Float16* Cb = C + (size_t)b * DD * DD;
  const int nloc = lane & 15;
  const int mhi = (lane & 16) ? 8 : 0;
#pragma unroll
  for (int tr = 0; tr < 2; ++tr)
#pragma unroll
    for (int tc = 0; tc < 4; ++tc) {
      int ng = c0 + tc * 16 + nloc;
#pragma unroll
      for (int r = 0; r < 8; ++r) {
        int mg = wave * 32 + tr * 16 + mhi + r;
        float v = acc[tr][tc][r];
        if (mode) v = 0.5f * (((mg == ng) ? 3.0f : 0.0f) - v);
        Cb[mg * DD + ng] = (_Float16)v;
      }
    }
}

// ---- K4: out[b,m] = sqrtNormA[b] * sum_e YZY[b,m,e]*mean[b,e] -------------
// Lane-parallel dots (coalesced f16 reads) + wave32 shuffle reduction.
__global__ __launch_bounds__(256) void matvec_kernel(
    const _Float16* __restrict__ YZY, const float* __restrict__ mean,
    const float* __restrict__ sqrtNormA, float* __restrict__ out) {
  const int b = blockIdx.x;
  const int tid = threadIdx.x;
  const int lane = tid & 31;
  const int wave = tid >> 5;
  __shared__ float sm[DD];
  sm[tid] = mean[b * DD + tid];
  __syncthreads();
  const float sc = sqrtNormA[b];
  const _Float16* base = YZY + (size_t)b * DD * DD;
  for (int r = 0; r < 32; ++r) {          // 8 waves x 32 rows = 256 rows
    const int m = wave * 32 + r;
    const _Float16* row = base + (size_t)m * DD;
    float acc = 0.f;
#pragma unroll
    for (int j = 0; j < DD / 32; ++j) {
      int e = lane + j * 32;
      acc += (float)row[e] * sm[e];
    }
#pragma unroll
    for (int off = 16; off > 0; off >>= 1) acc += __shfl_xor(acc, off, 32);
    if (lane == 0) out[b * DD + m] = sc * acc;
  }
}

extern "C" void kernel_launch(void* const* d_in, const int* in_sizes, int n_in,
                              void* d_out, int out_size, void* d_ws, size_t ws_size,
                              hipStream_t stream) {
  const float* feat  = (const float*)d_in[0];
  const float* noise = (const float*)d_in[1];
  float* out = (float*)d_out;

  char* ws = (char*)d_ws;
  size_t off = 0;
  auto carve = [&](size_t bytes) -> void* {
    void* p = ws + off;
    off += (bytes + 255) & ~(size_t)255;
    return p;
  };
  float* mean      = (float*)carve((size_t)NB * DD * sizeof(float));
  float* invTrace  = (float*)carve((size_t)NB * sizeof(float));
  float* sqrtNormA = (float*)carve((size_t)NB * sizeof(float));
  _Float16* Df = (_Float16*)carve((size_t)NB * NPG * DD * sizeof(_Float16)); // 32 MB
  _Float16* buf[4];
  for (int i = 0; i < 4; ++i)
    buf[i] = (_Float16*)carve((size_t)NB * DD * DD * sizeof(_Float16));      // 4x16 MB

  dim3 ggrid(4, NB);

  prep_kernel<<<NB, 256, 0, stream>>>(feat, noise, mean, Df, invTrace, sqrtNormA);
  // A -> buf0, T0 = 0.5(3I - A) -> buf1
  cov_kernel<<<ggrid, 256, 0, stream>>>(Df, invTrace, buf[0], buf[1]);
  // Y = A * T0 -> buf2 ; Z = T0 (buf1)
  ns_gemm_kernel<<<ggrid, 256, 0, stream>>>(buf[0], buf[1], buf[2], 0);

  _Float16 *pY = buf[2], *pZ = buf[1], *fa = buf[0], *fb = buf[3];
  for (int it = 0; it < ITERN - 2; ++it) {
    ns_gemm_kernel<<<ggrid, 256, 0, stream>>>(pZ, pY, fa, 1);  // T = 0.5(3I-Z*Y)
    ns_gemm_kernel<<<ggrid, 256, 0, stream>>>(pY, fa, fb, 0);  // Ynew = Y*T
    ns_gemm_kernel<<<ggrid, 256, 0, stream>>>(fa, pZ, pY, 0);  // Znew = T*Z
    _Float16* oldZ = pZ;
    pZ = pY;   // Znew
    pY = fb;   // Ynew
    fb = oldZ; // free
  }
  ns_gemm_kernel<<<ggrid, 256, 0, stream>>>(pZ, pY, fa, 1);    // T
  ns_gemm_kernel<<<ggrid, 256, 0, stream>>>(pY, fa, fb, 0);    // YZY = Y*T
  matvec_kernel<<<NB, 256, 0, stream>>>(fb, mean, sqrtNormA, out);
}
